// DocREModel_19284403159262
// MI455X (gfx1250) — matrix operand order repaired and checked by reading, not compiled, not run on hardware
//
#include <hip/hip_runtime.h>
#include <hip/hip_bf16.h>
#include <math.h>

#define B_   2
#define L_   512
#define D_   768
#define H_   12
#define E_   32
#define M_   4
#define P_   256
#define EMB_ 768
#define BS_  64
#define C_   97
#define N_   (B_*P_)    // 512
#define NM_  (E_*M_)    // 128
#define NKT_ 1536       // number of 32-wide K chunks in r1 (49152/32)

typedef __attribute__((ext_vector_type(16))) _Float16 v16h;
typedef __attribute__((ext_vector_type(8)))  _Float16 h8;
typedef __attribute__((ext_vector_type(8)))  float    v8f;
typedef __attribute__((ext_vector_type(4)))  float    f4;
typedef int i32x4 __attribute__((vector_size(16)));
struct F4x2 { f4 a, b; };
struct H8x2 { h8 a, b; };

// ---- CDNA5 async global->LDS path (ASYNCcnt), guarded so absence of the
// ---- builtin falls back to synchronous f16 staging.
#if defined(__HIP_DEVICE_COMPILE__) && __has_builtin(__builtin_amdgcn_global_load_async_to_lds_b128)
#define ASYNC_OK 1
__device__ __forceinline__ void async_b128(const _Float16* g, _Float16* l) {
  __builtin_amdgcn_global_load_async_to_lds_b128(
      (__attribute__((address_space(1))) i32x4*)g,
      (__attribute__((address_space(3))) i32x4*)l, 0, 0);
}
__device__ __forceinline__ void wait_async4() {
#if __has_builtin(__builtin_amdgcn_s_wait_asynccnt)
  __builtin_amdgcn_s_wait_asynccnt(4);
#else
  asm volatile("s_wait_asynccnt 0x4" ::: "memory");
#endif
}
__device__ __forceinline__ void wait_async0() {
#if __has_builtin(__builtin_amdgcn_s_wait_asynccnt)
  __builtin_amdgcn_s_wait_asynccnt(0);
#else
  asm volatile("s_wait_asynccnt 0x0" ::: "memory");
#endif
}
#else
#define ASYNC_OK 0
#endif

__device__ __forceinline__ float gelu_tanh(float x) {
  const float k0 = 0.7978845608028654f; // sqrt(2/pi)
  const float k1 = 0.044715f;
  return 0.5f * x * (1.0f + tanhf(k0 * (x + k1 * x * x * x)));
}

// -------- Stage 1: mention gather + logsumexp over M --------
__global__ void k_ment(const float* __restrict__ seq, const int* __restrict__ epos,
                       float* __restrict__ ment_flat, float* __restrict__ ent_emb) {
  int be = blockIdx.x;                 // b*E + e
  int b  = be / E_;
  int p0 = epos[be * M_ + 0] + 1;
  int p1 = epos[be * M_ + 1] + 1;
  int p2 = epos[be * M_ + 2] + 1;
  int p3 = epos[be * M_ + 3] + 1;
  const float* sb = seq + (long long)b * L_ * D_;
  int e = be % E_;
  for (int d = threadIdx.x; d < D_; d += blockDim.x) {
    float x0 = sb[(long long)p0 * D_ + d];
    float x1 = sb[(long long)p1 * D_ + d];
    float x2 = sb[(long long)p2 * D_ + d];
    float x3 = sb[(long long)p3 * D_ + d];
    long long mb = ((long long)b * NM_ + e * M_) * D_ + d;
    ment_flat[mb]          = x0;
    ment_flat[mb + D_]     = x1;
    ment_flat[mb + 2 * D_] = x2;
    ment_flat[mb + 3 * D_] = x3;
    float mx = fmaxf(fmaxf(x0, x1), fmaxf(x2, x3));
    float s  = expf(x0 - mx) + expf(x1 - mx) + expf(x2 - mx) + expf(x3 - mx);
    ent_emb[(long long)be * D_ + d] = mx + logf(s);
  }
}

// -------- Stage 2: ent_att[b,e,h,l] = mean_m attn[b,h,pos,l] --------
__global__ void k_entatt(const float* __restrict__ attn, const int* __restrict__ epos,
                         float* __restrict__ ent_att) {
  int bid = blockIdx.x;                // (b*E + e)*H + h
  int h   = bid % H_;
  int be  = bid / H_;
  int b   = be / E_;
  int p0 = epos[be * M_ + 0] + 1;
  int p1 = epos[be * M_ + 1] + 1;
  int p2 = epos[be * M_ + 2] + 1;
  int p3 = epos[be * M_ + 3] + 1;
  const float* ab = attn + (((long long)b * H_ + h) * L_) * L_;
  float* ob = ent_att + (long long)bid * L_;
  for (int l = threadIdx.x; l < L_; l += blockDim.x) {
    ob[l] = 0.25f * (ab[(long long)p0 * L_ + l] + ab[(long long)p1 * L_ + l] +
                     ab[(long long)p2 * L_ + l] + ab[(long long)p3 * L_ + l]);
  }
}

// -------- Stage 3: ht_att (product over H, normalize over L) --------
__global__ void k_htatt(const float* __restrict__ ent_att, const int* __restrict__ hts,
                        float* __restrict__ ht_att) {
  __shared__ float red[256];
  int n = blockIdx.x;                  // b*P + p
  int b = n / P_;
  int hi = hts[n * 2 + 0], ti = hts[n * 2 + 1];
  const float* ha = ent_att + (((long long)b * E_ + hi) * H_) * L_;
  const float* ta = ent_att + (((long long)b * E_ + ti) * H_) * L_;
  float v[2];
  float lsum = 0.f;
  #pragma unroll
  for (int t = 0; t < 2; t++) {
    int l = threadIdx.x + t * 256;
    float s = 0.f;
    #pragma unroll
    for (int h = 0; h < H_; h++) s += ha[h * L_ + l] * ta[h * L_ + l];
    s *= (1.0f / H_);
    v[t] = s;
    lsum += s;
  }
  red[threadIdx.x] = lsum;
  __syncthreads();
  for (int off = 128; off > 0; off >>= 1) {
    if (threadIdx.x < (unsigned)off) red[threadIdx.x] += red[threadIdx.x + off];
    __syncthreads();
  }
  float inv = 1.0f / (red[0] + 1e-5f);
  #pragma unroll
  for (int t = 0; t < 2; t++)
    ht_att[(long long)n * L_ + threadIdx.x + t * 256] = v[t] * inv;
}

// -------- Stage 5: build [hs|rs] and [ts|rs] A-matrices --------
__global__ void k_buildA(const float* __restrict__ ent_emb, const float* __restrict__ rs,
                         const int* __restrict__ hts, float* __restrict__ Ah,
                         float* __restrict__ At) {
  int n = blockIdx.x;
  int b = n / P_;
  int hi = hts[n * 2 + 0], ti = hts[n * 2 + 1];
  const float* eh = ent_emb + ((long long)b * E_ + hi) * D_;
  const float* et = ent_emb + ((long long)b * E_ + ti) * D_;
  const float* rn = rs + (long long)n * D_;
  float* ah = Ah + (long long)n * 2 * D_;
  float* at = At + (long long)n * 2 * D_;
  for (int d = threadIdx.x; d < D_; d += blockDim.x) {
    ah[d] = eh[d];
    at[d] = et[d];
    float r = rn[d];
    ah[D_ + d] = r;
    at[D_ + d] = r;
  }
}

// -------- Generic WMMA GEMM: C = act(A*B + bias), f32 in/out, f16 compute ----
// A: M x K (row major, lda), B: K x N (row major, ldb), C: M x N (ldc)
// Block = 128 threads (4 waves), tile 64x64, K step 32. Batched via blockIdx.z.
template <int ACT>
__global__ void gemm_wmma(const float* __restrict__ A, const float* __restrict__ B,
                          const float* __restrict__ bias, float* __restrict__ C,
                          int M, int K, int N, int lda, int ldb, int ldc,
                          long long sA, long long sB, long long sC) {
  __shared__ __align__(16) _Float16 At[64][32];  // row-major rows of 64B
  __shared__ __align__(16) _Float16 Bt[64][32];  // col-major: Bt[n][k]
  A += (long long)blockIdx.z * sA;
  B += (long long)blockIdx.z * sB;
  C += (long long)blockIdx.z * sC;
  int n0 = blockIdx.x * 64;
  int m0 = blockIdx.y * 64;
  int lane = threadIdx.x & 31;
  int wv   = threadIdx.x >> 5;
  int lrow = lane & 15;
  int hsel = lane >> 4;  // 0: lanes 0-15, 1: lanes 16-31

  v8f acc[4];
  #pragma unroll
  for (int t = 0; t < 4; t++)
    #pragma unroll
    for (int r = 0; r < 8; r++) acc[t][r] = 0.f;

  for (int k0 = 0; k0 < K; k0 += 32) {
    for (int i = threadIdx.x; i < 64 * 32; i += 128) {
      int m = i >> 5, kk = i & 31;
      int gm = m0 + m, gk = k0 + kk;
      float v = (gm < M && gk < K) ? A[(long long)gm * lda + gk] : 0.f;
      At[m][kk] = (_Float16)v;
    }
    for (int i = threadIdx.x; i < 64 * 32; i += 128) {
      int nn = i >> 5, kk = i & 31;
      int gn = n0 + nn, gk = k0 + kk;
      float v = (gn < N && gk < K) ? B[(long long)gk * ldb + gn] : 0.f;
      Bt[nn][kk] = (_Float16)v;
    }
    __syncthreads();
    // A fragment: lanes<16 take K[0:8)+K[16:24); lanes>=16 take K[8:16)+K[24:32)
    const f4* arow = reinterpret_cast<const f4*>(&At[wv * 16 + lrow][0]);
    F4x2 afr;
    afr.a = arow[hsel];
    afr.b = arow[hsel + 2];
    v16h av = __builtin_bit_cast(v16h, afr);
    #pragma unroll
    for (int t = 0; t < 4; t++) {
      // B fragment: lanes<16 take K[0:8)+K[16:24); lanes>=16 take K[8:16)+K[24:32) of their column
      const f4* bcol = reinterpret_cast<const f4*>(&Bt[t * 16 + lrow][0]);
      F4x2 bfr;
      bfr.a = bcol[hsel];
      bfr.b = bcol[hsel + 2];
      v16h bv = __builtin_bit_cast(v16h, bfr);
      acc[t] = __builtin_amdgcn_wmma_f32_16x16x32_f16(false, av, false, bv,
                                                      (short)0, acc[t], false, false);
    }
    __syncthreads();
  }

  #pragma unroll
  for (int t = 0; t < 4; t++) {
    #pragma unroll
    for (int r = 0; r < 8; r++) {
      int gm = m0 + wv * 16 + r + 8 * hsel;
      int gn = n0 + t * 16 + lrow;
      if (gm < M && gn < N) {
        float v = acc[t][r];
        if (bias) v += bias[gn];
        if (ACT == 1) v = tanhf(v);
        C[(long long)gm * ldc + gn] = v;
      }
    }
  }
}

// -------- pre-swizzle W_rel into f16 tile-contiguous layout ------------------
// Tile t (32 K-values): Wh[t*4096 + n*32 + kk] = f16(W_rel[(t*32+kk)*97 + n]),
// zero for n in [97,128). Each tile is one contiguous 8 KB block -> async DMA.
__global__ void k_wrel_pack(const float* __restrict__ Wrel, _Float16* __restrict__ Wh) {
  int t = blockIdx.x;                  // 0..1535
  _Float16* dst = Wh + (size_t)t * 4096;
  for (int i = threadIdx.x; i < 4096; i += 128) {
    int n = i >> 5, kk = i & 31;
    float v = (n < C_) ? Wrel[(size_t)(t * 32 + kk) * C_ + n] : 0.f;
    dst[i] = (_Float16)v;
  }
}

// -------- r1: (512 x 49152) x (49152 x 97), A generated in registers --------
// bl[n, g*4096 + a*64 + c] = hz[n, g*64+a] * tz[n, g*64+c]
// B tiles come from pre-swizzled Wh via double-buffered async loads to LDS.
__global__ void r1_wmma_kernel(const float* __restrict__ hz, const float* __restrict__ tz,
                               const _Float16* __restrict__ Wh, const float* __restrict__ brel,
                               float* __restrict__ r1) {
  __shared__ __align__(16) _Float16 Bt[2][128][32];  // double-buffered W_rel tiles (16 KB)
  __shared__ __align__(16) _Float16 Hseg[64][64];    // hz[row, g*64 + *]
  __shared__ __align__(16) _Float16 Tseg[64][64];    // tz[row, g*64 + *]
  int m0 = blockIdx.x * 64;
  int lane = threadIdx.x & 31;
  int wv   = threadIdx.x >> 5;
  int lrow = lane & 15;
  int hsel = lane >> 4;
  int rowl = wv * 16 + lrow;

  v8f acc[7];
  #pragma unroll
  for (int tt = 0; tt < 7; tt++)
    #pragma unroll
    for (int r = 0; r < 8; r++) acc[tt][r] = 0.f;

#if ASYNC_OK
  {  // prefetch tile 0 (4 x b128 per wave, 8 KB total)
    _Float16* ldst = &Bt[0][0][0];
    #pragma unroll
    for (int j = 0; j < 4; ++j) {
      int off = ((wv * 4 + j) * 32 + lane) * 8;  // f16 elements (16B chunks)
      async_b128(Wh + off, ldst + off);
    }
  }
#endif

  int t = 0;
  for (int g = 0; g < EMB_ / 64; ++g) {
    __syncthreads();  // prior A-fragment reads of Hseg/Tseg complete
    for (int i = threadIdx.x; i < 64 * 64; i += 128) {
      int row = i >> 6, c = i & 63;
      Hseg[row][c] = (_Float16)hz[(size_t)(m0 + row) * EMB_ + g * 64 + c];
      Tseg[row][c] = (_Float16)tz[(size_t)(m0 + row) * EMB_ + g * 64 + c];
    }
    for (int k0g = 0; k0g < 4096; k0g += 32, ++t) {
      __syncthreads();  // Hseg/Tseg visible; prior B-tile reads done before overwrite
#if ASYNC_OK
      bool has_next = (t + 1) < NKT_;
      if (has_next) {
        const _Float16* gsrc = Wh + (size_t)(t + 1) * 4096;
        _Float16* ldst = &Bt[(t + 1) & 1][0][0];
        #pragma unroll
        for (int j = 0; j < 4; ++j) {
          int off = ((wv * 4 + j) * 32 + lane) * 8;
          async_b128(gsrc + off, ldst + off);
        }
      }
#else
      {  // synchronous f16 staging fallback
        const f4* gsrc = (const f4*)(Wh + (size_t)t * 4096);
        f4* ldst = (f4*)&Bt[t & 1][0][0];
        for (int i = threadIdx.x; i < 512; i += 128) ldst[i] = gsrc[i];
      }
#endif
      // A fragment directly in registers: two contiguous 8-f16 runs of tz,
      // scaled by the hz scalar (packed f16 multiplies).
      int a  = k0g >> 6;
      int c0 = k0g & 63;  // 0 or 32
      _Float16 hf = Hseg[rowl][a];
      const h8* trow = (const h8*)&Tseg[rowl][0];  // 8 chunks of 8 f16
      H8x2 af;
      af.a = trow[(c0 >> 3) + hsel]     * hf;  // K[0:8) / K[8:16)
      af.b = trow[(c0 >> 3) + 2 + hsel] * hf;  // K[16:24) / K[24:32)
      v16h av = __builtin_bit_cast(v16h, af);
#if ASYNC_OK
      if (has_next) wait_async4(); else wait_async0();
#endif
      __syncthreads();  // B tile landed for all waves
      const _Float16(*bt)[32] = Bt[t & 1];
      #pragma unroll
      for (int tt = 0; tt < 7; tt++) {
        const f4* bcol = (const f4*)&bt[tt * 16 + lrow][0];
        F4x2 bfr;
        bfr.a = bcol[hsel];
        bfr.b = bcol[hsel + 2];
        v16h bv = __builtin_bit_cast(v16h, bfr);
        acc[tt] = __builtin_amdgcn_wmma_f32_16x16x32_f16(false, av, false, bv,
                                                         (short)0, acc[tt], false, false);
      }
    }
  }

  #pragma unroll
  for (int tt = 0; tt < 7; tt++) {
    #pragma unroll
    for (int r = 0; r < 8; r++) {
      int gm = m0 + wv * 16 + r + 8 * hsel;
      int gn = tt * 16 + lrow;
      if (gn < C_) r1[(long long)gm * C_ + gn] = acc[tt][r] + brel[gn];
    }
  }
}

// -------- constants: cu = b_f@W_unet + b_unet ; cm = b_f@W_mlp + b_mlp -------
__global__ void k_consts(const float* __restrict__ bf, const float* __restrict__ Wu,
                         const float* __restrict__ bu, const float* __restrict__ Wm,
                         const float* __restrict__ bm, float* __restrict__ cu,
                         float* __restrict__ cm) {
  int c = threadIdx.x;
  if (c < C_) {
    float s1 = bu[c], s2 = bm[c];
    for (int d = 0; d < D_; d++) {
      float v = bf[d];
      s1 += v * Wu[d * C_ + c];
      s2 += v * Wm[d * C_ + c];
    }
    cu[c] = s1;
    cm[c] = s2;
  }
}

// -------- r2/r3: 4x4 tanh/gelu + logsumexp over mention pairs ----------------
__global__ void k_r23(const float* __restrict__ u_h, const float* __restrict__ u_t,
                      const float* __restrict__ m_h, const float* __restrict__ m_t,
                      const float* __restrict__ cu, const float* __restrict__ cm,
                      const int* __restrict__ hts, const int* __restrict__ midx,
                      float* __restrict__ r2, float* __restrict__ r3) {
  int n = blockIdx.x;
  int b = n / P_;
  int hi = hts[n * 2 + 0], ti = hts[n * 2 + 1];
  int c = threadIdx.x;
  if (c >= C_) return;
  float cuc = cu[c], cmc = cm[c];
  float tv[16], gv[16];
  #pragma unroll
  for (int i = 0; i < M_; i++) {
    int mi = midx[((long long)b * E_ + hi) * M_ + i];
    float uh = u_h[((long long)b * NM_ + mi) * C_ + c];
    float mh = m_h[((long long)b * NM_ + mi) * C_ + c];
    #pragma unroll
    for (int j = 0; j < M_; j++) {
      int mj = midx[((long long)b * E_ + ti) * M_ + j];
      float ut = u_t[((long long)b * NM_ + mj) * C_ + c];
      float mt = m_t[((long long)b * NM_ + mj) * C_ + c];
      tv[i * 4 + j] = tanhf(uh + ut + cuc);
      gv[i * 4 + j] = gelu_tanh(mh + mt + cmc);
    }
  }
  float mx2 = tv[0], mx3 = gv[0];
  #pragma unroll
  for (int k = 1; k < 16; k++) {
    mx2 = fmaxf(mx2, tv[k]);
    mx3 = fmaxf(mx3, gv[k]);
  }
  float s2 = 0.f, s3 = 0.f;
  #pragma unroll
  for (int k = 0; k < 16; k++) {
    s2 += expf(tv[k] - mx2);
    s3 += expf(gv[k] - mx3);
  }
  r2[(long long)n * C_ + c] = mx2 + logf(s2);
  r3[(long long)n * C_ + c] = mx3 + logf(s3);
}

// -------- final: out = [r1|r2|r3] @ W_bil + b_bil ----------------------------
__global__ void k_final(const float* __restrict__ r1, const float* __restrict__ r2,
                        const float* __restrict__ r3, const float* __restrict__ Wb,
                        const float* __restrict__ bb, float* __restrict__ out) {
  int n = blockIdx.x;
  int c = threadIdx.x;
  if (c >= C_) return;
  float s = bb[c];
  for (int k = 0; k < C_; k++) s += r1[(long long)n * C_ + k] * Wb[k * C_ + c];
  for (int k = 0; k < C_; k++) s += r2[(long long)n * C_ + k] * Wb[(C_ + k) * C_ + c];
  for (int k = 0; k < C_; k++) s += r3[(long long)n * C_ + k] * Wb[(2 * C_ + k) * C_ + c];
  out[(long long)n * C_ + c] = s;
}

extern "C" void kernel_launch(void* const* d_in, const int* in_sizes, int n_in,
                              void* d_out, int out_size, void* d_ws, size_t ws_size,
                              hipStream_t stream) {
  (void)in_sizes; (void)n_in; (void)out_size; (void)ws_size;
  const float* seq    = (const float*)d_in[0];
  const float* attn   = (const float*)d_in[1];
  const int*   epos   = (const int*)d_in[2];
  const int*   hts    = (const int*)d_in[3];
  const int*   midx   = (const int*)d_in[4];
  const float* W_head = (const float*)d_in[5];
  const float* b_head = (const float*)d_in[6];
  const float* W_tail = (const float*)d_in[7];
  const float* b_tail = (const float*)d_in[8];
  const float* W_rel  = (const float*)d_in[9];
  const float* b_rel  = (const float*)d_in[10];
  const float* W_fh   = (const float*)d_in[11];
  const float* W_ft   = (const float*)d_in[12];
  const float* b_f    = (const float*)d_in[13];
  const float* W_unet = (const float*)d_in[14];
  const float* b_unet = (const float*)d_in[15];
  const float* W_mlp  = (const float*)d_in[16];
  const float* b_mlp  = (const float*)d_in[17];
  const float* W_bil  = (const float*)d_in[18];
  const float* b_bil  = (const float*)d_in[19];

  float* ws = (float*)d_ws;
  size_t off = 0;
  float* ment_flat = ws + off; off += (size_t)B_ * NM_ * D_;     // 196608
  float* ent_emb   = ws + off; off += (size_t)B_ * E_ * D_;      // 49152
  float* ent_att   = ws + off; off += (size_t)B_ * E_ * H_ * L_; // 393216
  float* ht_att    = ws + off; off += (size_t)B_ * P_ * L_;      // 262144
  float* rs        = ws + off; off += (size_t)N_ * D_;           // 393216
  float* Ah        = ws + off; off += (size_t)N_ * 2 * D_;       // 786432
  float* At_      = ws + off; off += (size_t)N_ * 2 * D_;        // 786432
  float* hz        = ws + off; off += (size_t)N_ * EMB_;         // 393216
  float* tz        = ws + off; off += (size_t)N_ * EMB_;         // 393216
  float* proj_h    = ws + off; off += (size_t)B_ * NM_ * EMB_;   // 196608
  float* proj_t    = ws + off; off += (size_t)B_ * NM_ * EMB_;   // 196608
  float* u_h       = ws + off; off += (size_t)B_ * NM_ * C_;
  float* u_t       = ws + off; off += (size_t)B_ * NM_ * C_;
  float* m_h       = ws + off; off += (size_t)B_ * NM_ * C_;
  float* m_t       = ws + off; off += (size_t)B_ * NM_ * C_;
  float* cu        = ws + off; off += 128;
  float* cm        = ws + off; off += 128;
  float* r1        = ws + off; off += (size_t)N_ * C_;
  float* r2        = ws + off; off += (size_t)N_ * C_;
  float* r3        = ws + off; off += (size_t)N_ * C_;
  _Float16* Wh     = (_Float16*)(ws + off); off += (size_t)NKT_ * 4096 / 2; // 12.6 MB f16

  // Stage 0: pre-swizzle W_rel to f16 tiles (amortized; enables async B tiles)
  k_wrel_pack<<<NKT_, 128, 0, stream>>>(W_rel, Wh);
  // Stage 1: mention embeddings + entity logsumexp
  k_ment<<<B_ * E_, 256, 0, stream>>>(seq, epos, ment_flat, ent_emb);
  // Stage 2: entity attention (mean over mentions)
  k_entatt<<<B_ * E_ * H_, 256, 0, stream>>>(attn, epos, ent_att);
  // Stage 3: pair attention, normalized
  k_htatt<<<N_, 256, 0, stream>>>(ent_att, hts, ht_att);
  // Stage 4: rs = ht_att @ seq  (batched: 256x512x768 per batch)
  gemm_wmma<0><<<dim3(D_ / 64, P_ / 64, B_), 128, 0, stream>>>(
      ht_att, seq, nullptr, rs, P_, L_, D_, L_, D_, D_,
      (long long)P_ * L_, (long long)L_ * D_, (long long)P_ * D_);
  // Stage 5: build [hs|rs], [ts|rs]
  k_buildA<<<N_, 256, 0, stream>>>(ent_emb, rs, hts, Ah, At_);
  // Stage 6: hz/tz = tanh(A @ W + b)   (512x1536x768)
  gemm_wmma<1><<<dim3(D_ / 64, N_ / 64, 1), 128, 0, stream>>>(
      Ah, W_head, b_head, hz, N_, 2 * D_, EMB_, 2 * D_, EMB_, EMB_, 0, 0, 0);
  gemm_wmma<1><<<dim3(D_ / 64, N_ / 64, 1), 128, 0, stream>>>(
      At_, W_tail, b_tail, tz, N_, 2 * D_, EMB_, 2 * D_, EMB_, EMB_, 0, 0, 0);
  // Stage 7: r1 = bl @ W_rel + b_rel  (512x49152x97, A in registers, async B)
  r1_wmma_kernel<<<N_ / 64, 128, 0, stream>>>(hz, tz, Wh, b_rel, r1);
  // Stage 8: mention projections (256x768x768) then channel projections (256x768x97)
  gemm_wmma<0><<<dim3(D_ / 64, (B_ * NM_) / 64, 1), 128, 0, stream>>>(
      ment_flat, W_fh, nullptr, proj_h, B_ * NM_, D_, EMB_, D_, EMB_, EMB_, 0, 0, 0);
  gemm_wmma<0><<<dim3(D_ / 64, (B_ * NM_) / 64, 1), 128, 0, stream>>>(
      ment_flat, W_ft, nullptr, proj_t, B_ * NM_, D_, EMB_, D_, EMB_, EMB_, 0, 0, 0);
  gemm_wmma<0><<<dim3(2, (B_ * NM_) / 64, 1), 128, 0, stream>>>(
      proj_h, W_unet, nullptr, u_h, B_ * NM_, EMB_, C_, EMB_, C_, C_, 0, 0, 0);
  gemm_wmma<0><<<dim3(2, (B_ * NM_) / 64, 1), 128, 0, stream>>>(
      proj_t, W_unet, nullptr, u_t, B_ * NM_, EMB_, C_, EMB_, C_, C_, 0, 0, 0);
  gemm_wmma<0><<<dim3(2, (B_ * NM_) / 64, 1), 128, 0, stream>>>(
      proj_h, W_mlp, nullptr, m_h, B_ * NM_, EMB_, C_, EMB_, C_, C_, 0, 0, 0);
  gemm_wmma<0><<<dim3(2, (B_ * NM_) / 64, 1), 128, 0, stream>>>(
      proj_t, W_mlp, nullptr, m_t, B_ * NM_, EMB_, C_, EMB_, C_, C_, 0, 0, 0);
  // Stage 9: bias constants through W_unet / W_mlp
  k_consts<<<1, 128, 0, stream>>>(b_f, W_unet, b_unet, W_mlp, b_mlp, cu, cm);
  // Stage 10: r2/r3 pointwise logsumexp over 4x4 mention pairs
  k_r23<<<N_, 128, 0, stream>>>(u_h, u_t, m_h, m_t, cu, cm, hts, midx, r2, r3);
  // Stage 11: final projection
  k_final<<<N_, 128, 0, stream>>>(r1, r2, r3, W_bil, b_bil, (float*)d_out);
}
